// CasualAttention_37847251812797
// MI455X (gfx1250) — matrix operand order repaired and checked by reading, not compiled, not run on hardware
//
#include <hip/hip_runtime.h>

// ---------------------------------------------------------------------------
// CDNA5 (gfx1250) causal attention, bf16 WMMA compute end-to-end:
//   0) convert x, w_attn, w_proj fp32 -> bf16 (one pass, out of hot loops)
//   1) qkv_bf = x_bf @ w_attn_bf + b_attn   (WMMA; double-buffered async-LDS)
//   2) y_bf   = flash-attention(qkv_bf)     (WMMA; double-buffered TDM K tiles)
//   3) out    = y_bf @ w_proj_bf + b_proj   (WMMA, fp32 output)
// ---------------------------------------------------------------------------

typedef __attribute__((ext_vector_type(16))) __bf16    bf16x16;
typedef __attribute__((ext_vector_type(8)))  __bf16    bf16x8;
typedef __attribute__((ext_vector_type(4)))  __bf16    bf16x4;
typedef __attribute__((ext_vector_type(8)))  float     v8f;
typedef __attribute__((ext_vector_type(4)))  unsigned  uint4v;
typedef __attribute__((ext_vector_type(8)))  unsigned  uint8v;

static constexpr int D_MODEL  = 1024;
static constexpr int N_HEADS  = 16;
static constexpr int HEAD_DIM = 64;
static constexpr int SEQ      = 2048;
static constexpr int BATCH    = 4;
static constexpr int ROWS     = BATCH * SEQ;   // 8192

__device__ __forceinline__ bf16x16 cat8(bf16x8 lo, bf16x8 hi) {
  return __builtin_shufflevector(lo, hi, 0,1,2,3,4,5,6,7,8,9,10,11,12,13,14,15);
}

__device__ __forceinline__ v8f wmma_bf16(bf16x16 a, bf16x16 b, v8f c) {
  return __builtin_amdgcn_wmma_f32_16x16x32_bf16(
      false, a, false, b, (short)0, c, false, false);
}

// Low 32 bits of a generic pointer to LDS == LDS byte address (ISA 10.2).
__device__ __forceinline__ unsigned lds_addr_of(const void* p) {
  return (unsigned)(unsigned long long)p;
}

// ASYNCcnt-tracked 16B global->LDS copy (no VGPR data), GV addressing.
__device__ __forceinline__ void async_copy_b128(unsigned lds_byte, const void* gaddr) {
  asm volatile("global_load_async_to_lds_b128 %0, %1, off"
               :: "v"(lds_byte), "v"(gaddr) : "memory");
}
__device__ __forceinline__ void wait_async0() {
  asm volatile("s_wait_asynccnt 0x0" ::: "memory");
}
__device__ __forceinline__ void wait_async2() {  // allow the 2 copies of the next tile in flight
  asm volatile("s_wait_asynccnt 0x2" ::: "memory");
}
__device__ __forceinline__ void wait_tensor0() {
  asm volatile("s_wait_tensorcnt 0x0" ::: "memory");
}
__device__ __forceinline__ void wait_tensor1() {  // allow next tile's TDM op in flight
  asm volatile("s_wait_tensorcnt 0x1" ::: "memory");
}

// ---------------------------------------------------------------------------
// 0) fp32 -> bf16 elementwise (n multiple of 1024)
// ---------------------------------------------------------------------------
__global__ __launch_bounds__(256)
void f32_to_bf16_kernel(const float* __restrict__ in, __bf16* __restrict__ out, long n) {
  long i = ((long)blockIdx.x * 256 + threadIdx.x) * 4;
  if (i < n) {
    float4 f = *(const float4*)(in + i);
    bf16x4 o4 = { (__bf16)f.x, (__bf16)f.y, (__bf16)f.z, (__bf16)f.w };
    *(bf16x4*)(out + i) = o4;
  }
}

// ---------------------------------------------------------------------------
// GEMM: C[M,N] = A[M,K]*B[K,N] + bias[N], bf16 operands, fp32 accumulate.
// Block 128x128x32, 256 threads (8 waves), wave tile 64x32.
// Double-buffered LDS: tile t+1 staged (async A copies + B transpose) while
// tile t's 8 WMMAs run.
// ---------------------------------------------------------------------------
#define G_BM 128
#define G_BN 128
#define G_BK 32
#define G_LDS (G_BK + 8)   // padded row stride (elements): 80 bytes

template <bool OUT_BF16>
__global__ __launch_bounds__(256)
void gemm_bf16_kernel(const __bf16* __restrict__ A,
                      const __bf16* __restrict__ B,
                      const float* __restrict__ bias,
                      void* __restrict__ Cout,
                      int M, int K, int N) {
  __shared__ __attribute__((aligned(16))) __bf16 sA[2][G_BM * G_LDS];  // [m][k]
  __shared__ __attribute__((aligned(16))) __bf16 sB[2][G_BN * G_LDS];  // [n][k]

  const int tid  = threadIdx.x;
  const int lane = tid & 31;
  const int wv   = tid >> 5;
  const int half = lane >> 4;
  const int l16  = lane & 15;
  const int wm   = (wv >> 2) * 64;
  const int wn   = (wv & 3) * 32;
  const long bm  = (long)blockIdx.y * G_BM;
  const long bn  = (long)blockIdx.x * G_BN;

  auto stage = [&](int k0, int buf) {
    // A tile 128x32 bf16 = 512 x 16B chunks; 2 async copies per thread.
    const unsigned sAb = lds_addr_of(&sA[buf][0]);
#pragma unroll
    for (int i = 0; i < 2; ++i) {
      int idx = tid + i * 256;
      int r  = idx >> 2;          // 0..127
      int ce = (idx & 3) * 8;     // 0,8,16,24
      async_copy_b128(sAb + (unsigned)(r * G_LDS + ce) * 2,
                      A + (bm + r) * (long)K + k0 + ce);
    }
    // B tile 32x128 bf16, transpose-store to [n][k].
#pragma unroll
    for (int i = 0; i < 2; ++i) {
      int idx = tid + i * 256;
      int r  = idx >> 4;          // 0..31  (k)
      int ce = (idx & 15) * 8;    // 0..120 (n)
      bf16x8 v = *(const bf16x8*)(B + (long)(k0 + r) * N + bn + ce);
#pragma unroll
      for (int e = 0; e < 8; ++e) sB[buf][(ce + e) * G_LDS + r] = v[e];
    }
  };

  v8f acc[4][2];
#pragma unroll
  for (int mt = 0; mt < 4; ++mt)
#pragma unroll
    for (int nt = 0; nt < 2; ++nt)
      acc[mt][nt] = (v8f){0.f,0.f,0.f,0.f,0.f,0.f,0.f,0.f};

  const int nT = K / G_BK;
  stage(0, 0);
  for (int t = 0; t < nT; ++t) {
    const int cur = t & 1;
    if (t + 1 < nT) { stage((t + 1) * G_BK, cur ^ 1); wait_async2(); }
    else            { wait_async0(); }
    __syncthreads();   // staging of tile t visible to all waves

    bf16x16 af[4];
#pragma unroll
    for (int mt = 0; mt < 4; ++mt) {
      int m = wm + mt * 16 + l16;
      bf16x8 a0 = *(const bf16x8*)&sA[cur][m * G_LDS + half * 8];
      bf16x8 a1 = *(const bf16x8*)&sA[cur][m * G_LDS + 16 + half * 8];
      af[mt] = cat8(a0, a1);
    }
    bf16x16 bf[2];
#pragma unroll
    for (int nt = 0; nt < 2; ++nt) {
      int n = wn + nt * 16 + l16;
      bf16x8 b0 = *(const bf16x8*)&sB[cur][n * G_LDS + half * 8];
      bf16x8 b1 = *(const bf16x8*)&sB[cur][n * G_LDS + 16 + half * 8];
      bf[nt] = cat8(b0, b1);
    }
#pragma unroll
    for (int mt = 0; mt < 4; ++mt)
#pragma unroll
      for (int nt = 0; nt < 2; ++nt)
        acc[mt][nt] = wmma_bf16(af[mt], bf[nt], acc[mt][nt]);
    __syncthreads();   // reads done before buffer 'cur' is restaged at t+2
  }

  // Epilogue (C layout: M = e + 8*half, N = l16).
#pragma unroll
  for (int mt = 0; mt < 4; ++mt) {
#pragma unroll
    for (int nt = 0; nt < 2; ++nt) {
      long n = bn + wn + nt * 16 + l16;
      float bv = bias[n];
#pragma unroll
      for (int e = 0; e < 8; ++e) {
        long m = bm + wm + mt * 16 + e + half * 8;
        float r = acc[mt][nt][e] + bv;
        if (OUT_BF16) ((__bf16*)Cout)[m * (long)N + n] = (__bf16)r;
        else          ((float*)Cout)[m * (long)N + n] = r;
      }
    }
  }
}

// ---------------------------------------------------------------------------
// Flash attention.  128 threads (4 waves) per (b, h, 64-row q tile).
// K tiles staged by the Tensor Data Mover (wave 0 issues the descriptor for
// tile j+1 before computing tile j; TENSORcnt<=1 pipelines the DMA).
// V transposed manually (double-buffered), online softmax in fp32.
// ---------------------------------------------------------------------------
#define A_LDSW 72   // padded row stride: 144B = 128B data + 16B pad (TDM pad)

__global__ __launch_bounds__(128)
void attn_kernel(const __bf16* __restrict__ qkv, __bf16* __restrict__ y) {
  __shared__ __attribute__((aligned(16))) __bf16 sK[2][64 * A_LDSW];  // [kv][dh]
  __shared__ __attribute__((aligned(16))) __bf16 sV[2][64 * A_LDSW];  // [dh][kv]
  __shared__ __attribute__((aligned(16))) __bf16 sP[64 * A_LDSW];     // [qrow][kv]

  const int qt   = blockIdx.x;
  const int bh   = blockIdx.y;
  const int b    = bh >> 4;
  const int h    = bh & 15;
  const int tid  = threadIdx.x;
  const int lane = tid & 31;
  const int wv   = tid >> 5;
  const int half = lane >> 4;
  const int l16  = lane & 15;

  const long rowBase = (long)b * SEQ;
  const long ldq = 3 * D_MODEL;
  const float scale = 0.125f;   // 1/sqrt(HEAD_DIM), applied to fp32 scores

  // K tile via TDM: 64x64 bf16 2-D tile, row stride 3072 elements,
  // LDS padding 4 dwords after every 32 dwords (-> 144B rows == A_LDSW).
  auto stageK_tdm = [&](int j, int buf) {
    if (wv != 0) return;
    const __bf16* kp = qkv + (rowBase + j * 64) * ldq + D_MODEL + h * HEAD_DIM;
    unsigned long long ga = (unsigned long long)(const void*)kp;
    uint4v g0;
    g0[0] = 1u;                                   // count=1, user mode
    g0[1] = lds_addr_of(&sK[buf][0]);             // lds_addr
    g0[2] = (unsigned)ga;                         // global_addr[31:0]
    g0[3] = (unsigned)(ga >> 32) | (2u << 30);    // global_addr[56:32] | type=2
    const unsigned td0 = 64, td1 = 64, tile0 = 64, tile1 = 64;
    const unsigned long long st0 = 3 * D_MODEL;   // tensor_dim0_stride
    uint8v g1;
    g1[0] = (1u << 16)        // data_size = 2 bytes
          | (1u << 20)        // pad_enable
          | (4u << 22)        // pad_interval: every 32 dwords
          | (3u << 25);       // pad_amount: 4 dwords
    g1[1] = (td0 & 0xFFFFu) << 16;                     // dim0[15:0]
    g1[2] = (td0 >> 16) | ((td1 & 0xFFFFu) << 16);     // dim0[31:16] | dim1[15:0]
    g1[3] = (td1 >> 16) | (tile0 << 16);               // dim1[31:16] | tile_dim0
    g1[4] = tile1;                                     // tile_dim1 | tile_dim2=0
    g1[5] = (unsigned)st0;                             // stride0[31:0]
    g1[6] = (unsigned)(st0 >> 32) & 0xFFFFu;           // stride0[47:32]
    g1[7] = 0u;
    asm volatile("tensor_load_to_lds %0, %1" :: "s"(g0), "s"(g1) : "memory");
  };

  // V tile transposed to [dh][kv] (512 x 16B chunks / 128 threads).
  auto stageV = [&](int j, int buf) {
    const __bf16* vp = qkv + (rowBase + j * 64) * ldq + 2 * D_MODEL + h * HEAD_DIM;
#pragma unroll
    for (int i = 0; i < 4; ++i) {
      int idx = tid + i * 128;
      int r  = idx >> 3;          // 0..63 (kv row)
      int ce = (idx & 7) * 8;     // 0..56 (dh col)
      bf16x8 v = *(const bf16x8*)(vp + (long)r * ldq + ce);
#pragma unroll
      for (int e = 0; e < 8; ++e) sV[buf][(ce + e) * A_LDSW + r] = v[e];
    }
  };

  // Q fragments in registers (A layout, M = l16).
  bf16x16 qf[2];
  {
    const __bf16* qp = qkv + (rowBase + qt * 64 + wv * 16 + l16) * ldq + h * HEAD_DIM;
#pragma unroll
    for (int ks = 0; ks < 2; ++ks) {
      bf16x8 a0 = *(const bf16x8*)(qp + ks * 32 + half * 8);
      bf16x8 a1 = *(const bf16x8*)(qp + ks * 32 + 16 + half * 8);
      qf[ks] = cat8(a0, a1);
    }
  }

  float mrow[8], lrow[8];
  v8f o[4];
#pragma unroll
  for (int e = 0; e < 8; ++e) { mrow[e] = -__builtin_inff(); lrow[e] = 0.f; }
#pragma unroll
  for (int dt = 0; dt < 4; ++dt) o[dt] = (v8f){0.f,0.f,0.f,0.f,0.f,0.f,0.f,0.f};

  stageK_tdm(0, 0);
  stageV(0, 0);
  for (int j = 0; j <= qt; ++j) {
    const int cur = j & 1;
    if (j < qt) {                      // pipeline next kv tile
      stageK_tdm(j + 1, cur ^ 1);
      stageV(j + 1, cur ^ 1);
      if (wv == 0) wait_tensor1();     // tile j's TDM done (in-order)
    } else {
      if (wv == 0) wait_tensor0();
    }
    __syncthreads();   // K (TDM) + V staging of tile j visible to all waves

    // S = Q * K^T
    v8f s[4];
#pragma unroll
    for (int nt = 0; nt < 4; ++nt) s[nt] = (v8f){0.f,0.f,0.f,0.f,0.f,0.f,0.f,0.f};
#pragma unroll
    for (int ks = 0; ks < 2; ++ks) {
#pragma unroll
      for (int nt = 0; nt < 4; ++nt) {
        int n = nt * 16 + l16;
        bf16x8 b0 = *(const bf16x8*)&sK[cur][n * A_LDSW + ks * 32 + half * 8];
        bf16x8 b1 = *(const bf16x8*)&sK[cur][n * A_LDSW + ks * 32 + 16 + half * 8];
        s[nt] = wmma_bf16(qf[ks], cat8(b0, b1), s[nt]);
      }
    }
#pragma unroll
    for (int nt = 0; nt < 4; ++nt) s[nt] = s[nt] * scale;

    if (j == qt) {   // causal mask on the diagonal tile
#pragma unroll
      for (int nt = 0; nt < 4; ++nt) {
        int kcol = nt * 16 + l16;
#pragma unroll
        for (int e = 0; e < 8; ++e) {
          int qrow = wv * 16 + e + half * 8;
          if (kcol > qrow) s[nt][e] = -__builtin_inff();
        }
      }
    }

    // Online softmax (row = e + 8*half; reduce across 16 lanes of the half).
#pragma unroll
    for (int e = 0; e < 8; ++e) {
      float v = fmaxf(fmaxf(s[0][e], s[1][e]), fmaxf(s[2][e], s[3][e]));
#pragma unroll
      for (int m = 8; m >= 1; m >>= 1) v = fmaxf(v, __shfl_xor(v, m, 32));
      float mn = fmaxf(mrow[e], v);
      float alpha = __expf(mrow[e] - mn);
      mrow[e] = mn;
      float ps = 0.f;
#pragma unroll
      for (int nt = 0; nt < 4; ++nt) {
        float p = __expf(s[nt][e] - mn);
        s[nt][e] = p;
        ps += p;
      }
#pragma unroll
      for (int m = 8; m >= 1; m >>= 1) ps += __shfl_xor(ps, m, 32);
      lrow[e] = lrow[e] * alpha + ps;
#pragma unroll
      for (int dt = 0; dt < 4; ++dt) o[dt][e] *= alpha;
    }

    // P: C-layout regs -> bf16 LDS (wave-private rows).
#pragma unroll
    for (int nt = 0; nt < 4; ++nt)
#pragma unroll
      for (int e = 0; e < 8; ++e)
        sP[(wv * 16 + e + half * 8) * A_LDSW + nt * 16 + l16] = (__bf16)s[nt][e];

    // O += P * V
#pragma unroll
    for (int ks = 0; ks < 2; ++ks) {
      const __bf16* pr = &sP[(wv * 16 + l16) * A_LDSW];
      bf16x8 a0 = *(const bf16x8*)&pr[ks * 32 + half * 8];
      bf16x8 a1 = *(const bf16x8*)&pr[ks * 32 + 16 + half * 8];
      bf16x16 pf = cat8(a0, a1);
#pragma unroll
      for (int dt = 0; dt < 4; ++dt) {
        int n = dt * 16 + l16;
        bf16x8 b0 = *(const bf16x8*)&sV[cur][n * A_LDSW + ks * 32 + half * 8];
        bf16x8 b1 = *(const bf16x8*)&sV[cur][n * A_LDSW + ks * 32 + 16 + half * 8];
        o[dt] = wmma_bf16(pf, cat8(b0, b1), o[dt]);
      }
    }
    __syncthreads();   // reads of tile j done before buffer 'cur' restaged at j+2
  }

  // Normalize, write bf16 y in [B*T, D_MODEL] layout.
#pragma unroll
  for (int e = 0; e < 8; ++e) {
    float inv = 1.0f / lrow[e];
    long row = rowBase + qt * 64 + wv * 16 + e + half * 8;
#pragma unroll
    for (int dt = 0; dt < 4; ++dt) {
      int dh = dt * 16 + l16;
      y[row * D_MODEL + h * HEAD_DIM + dh] = (__bf16)(o[dt][e] * inv);
    }
  }
}

// ---------------------------------------------------------------------------
extern "C" void kernel_launch(void* const* d_in, const int* in_sizes, int n_in,
                              void* d_out, int out_size, void* d_ws, size_t ws_size,
                              hipStream_t stream) {
  const float* x      = (const float*)d_in[0];
  const float* w_attn = (const float*)d_in[1];
  const float* b_attn = (const float*)d_in[2];
  const float* w_proj = (const float*)d_in[3];
  const float* b_proj = (const float*)d_in[4];
  float* out = (float*)d_out;

  // bf16 workspace
  __bf16* xb   = (__bf16*)d_ws;                          // 8192x1024
  __bf16* wab  = xb   + (size_t)ROWS * D_MODEL;          // 1024x3072
  __bf16* wpb  = wab  + (size_t)D_MODEL * 3 * D_MODEL;   // 1024x1024
  __bf16* qkvb = wpb  + (size_t)D_MODEL * D_MODEL;       // 8192x3072
  __bf16* yb   = qkvb + (size_t)ROWS * 3 * D_MODEL;      // 8192x1024

  const long nx  = (long)ROWS * D_MODEL;
  const long nwa = (long)D_MODEL * 3 * D_MODEL;
  const long nwp = (long)D_MODEL * D_MODEL;
  f32_to_bf16_kernel<<<dim3(nx  / 1024), 256, 0, stream>>>(x,      xb,  nx);
  f32_to_bf16_kernel<<<dim3(nwa / 1024), 256, 0, stream>>>(w_attn, wab, nwa);
  f32_to_bf16_kernel<<<dim3(nwp / 1024), 256, 0, stream>>>(w_proj, wpb, nwp);

  dim3 g1(3 * D_MODEL / G_BN, ROWS / G_BM);
  gemm_bf16_kernel<true><<<g1, 256, 0, stream>>>(xb, wab, b_attn, qkvb,
                                                 ROWS, D_MODEL, 3 * D_MODEL);

  dim3 g2(SEQ / 64, BATCH * N_HEADS);
  attn_kernel<<<g2, 128, 0, stream>>>(qkvb, yb);

  dim3 g3(D_MODEL / G_BN, ROWS / G_BM);
  gemm_bf16_kernel<false><<<g3, 256, 0, stream>>>(yb, wpb, b_proj, out,
                                                  ROWS, D_MODEL, D_MODEL);
}
